// crcab_block_13864154431927
// MI455X (gfx1250) — matrix operand order repaired and checked by reading, not compiled, not run on hardware
//
#include <hip/hip_runtime.h>
#include <hip/hip_bf16.h>

typedef __attribute__((ext_vector_type(16))) _Float16 v16h;
typedef __attribute__((ext_vector_type(8)))  _Float16 v8h;
typedef __attribute__((ext_vector_type(8)))  float    v8f;

union V16 { v16h v; v8h h[2]; };

#define CC   32
#define HH   64
#define WW   64
#define KD   288          // C*9 contraction depth
#define O2   (CC*KD)      // 9216 conv2 output channels
#define TP   16           // pixels per tile (WMMA N)
#define MT   18           // M tiles per output channel (288/16)
#define KKS  9            // k-steps (288/32)
#define LDH  296          // padded LDS stride (halves)
#define LDX  296          // padded LDS stride (floats)

// Branchless tanh: prefer the CDNA5 hardware transcendental (v_tanh_f32),
// fall back to a 5-op exp2/rcp approximation (no EXEC-mask slow path).
__device__ __forceinline__ float fast_tanh(float v) {
#if __has_builtin(__builtin_amdgcn_tanhf)
    return __builtin_amdgcn_tanhf(v);
#elif __has_builtin(__builtin_amdgcn_tanh_f32)
    return __builtin_amdgcn_tanh_f32(v);
#else
    float a = __builtin_fabsf(v);
    // e = exp(-2a) = exp2(-2*log2(e)*a); a>=0 so e in (0,1], no overflow.
    float e = __builtin_amdgcn_exp2f(a * -2.885390081777927f);
    float r = (1.0f - e) * __builtin_amdgcn_rcpf(1.0f + e);
    return __builtin_copysignf(r, v);
#endif
}

// ---------------------------------------------------------------------------
// Kernel 0: convert conv2_w fp32 -> f16 row-major [9216][288] in workspace
// ---------------------------------------------------------------------------
__global__ __launch_bounds__(256) void k_cvt_w2(const float* __restrict__ w2,
                                                _Float16* __restrict__ w2h, int n) {
    for (int i = blockIdx.x * blockDim.x + threadIdx.x; i < n; i += gridDim.x * blockDim.x)
        w2h[i] = (_Float16)w2[i];
}

// ---------------------------------------------------------------------------
// Kernel 1: conv1 (3x3, C->C, pad 1) + ReLU, emit h as f16 [N][C][H][W]
// One block per (n, y): 256 threads compute 32 channels x 64 x's.
// ---------------------------------------------------------------------------
__global__ __launch_bounds__(256) void k_conv1(const float* __restrict__ x,
                                               const float* __restrict__ w1,
                                               _Float16* __restrict__ hbuf) {
    __shared__ float xs[CC * 3 * 66];     // [i][kh][xx(0..65)], halo columns
    __shared__ float ws[CC * KD];         // conv1_w flat: [c][i*9+kh*3+kw]

    const int n = blockIdx.x / HH;
    const int y = blockIdx.x % HH;
    const int tid = threadIdx.x;

    for (int e = tid; e < CC * 3 * 66; e += 256) {
        int i  = e / 198;
        int r  = e - i * 198;
        int kh = r / 66;
        int xx = r - kh * 66;
        int ysrc = y + kh - 1;
        int xsrc = xx - 1;
        float v = 0.f;
        if ((unsigned)ysrc < HH && (unsigned)xsrc < WW)
            v = x[(((size_t)n * CC + i) * HH + ysrc) * WW + xsrc];
        xs[e] = v;
    }
    for (int e = tid; e < CC * KD; e += 256) ws[e] = w1[e];
    __syncthreads();

    const int xloc  = tid & 63;
    const int cbase = (tid >> 6) * 8;
    float acc[8];
#pragma unroll
    for (int j = 0; j < 8; ++j) acc[j] = 0.f;

    for (int i = 0; i < CC; ++i) {
#pragma unroll
        for (int kh = 0; kh < 3; ++kh) {
#pragma unroll
            for (int kw = 0; kw < 3; ++kw) {
                float v = xs[(i * 3 + kh) * 66 + xloc + kw];
                int k = i * 9 + kh * 3 + kw;
#pragma unroll
                for (int j = 0; j < 8; ++j)
                    acc[j] = fmaf(ws[(cbase + j) * KD + k], v, acc[j]);
            }
        }
    }
#pragma unroll
    for (int j = 0; j < 8; ++j)
        hbuf[(((size_t)n * CC + cbase + j) * HH + y) * WW + xloc] =
            (_Float16)fmaxf(acc[j], 0.f);
}

// ---------------------------------------------------------------------------
// Kernel 2: fused conv2 + tanh + per-pixel dynamic conv contraction.
// Block = 16-pixel row segment; 8 waves x 4 output channels each.
// ---------------------------------------------------------------------------
__global__ __launch_bounds__(256, 1) void k_main(const float* __restrict__ x,
                                                 const _Float16* __restrict__ w2h,
                                                 const _Float16* __restrict__ hbuf,
                                                 float* __restrict__ out) {
    __shared__ _Float16 hT[TP * LDH];   // h patches, pixel-major [p][k]
    __shared__ float    xT[TP * LDX];   // x patches, pixel-major [p][k]

    const int tprow = WW / TP;                       // 4 tiles per row
    const int b  = blockIdx.x;
    const int n  = b / (HH * tprow);
    const int rm = b % (HH * tprow);
    const int y  = rm / tprow;
    const int x0 = (rm % tprow) * TP;
    const int tid = threadIdx.x;

    // Stage patch matrices into LDS (k = i*9 + kh*3 + kw, zero-pad borders)
    for (int e = tid; e < TP * KD; e += 256) {
        int p  = e / KD;
        int k  = e - p * KD;
        int i2 = k / 9;
        int r  = k - i2 * 9;
        int kh = r / 3;
        int kw = r - kh * 3;
        int ys = y + kh - 1;
        int xsrc = x0 + p + kw - 1;
        bool ok = ((unsigned)ys < HH) & ((unsigned)xsrc < WW);
        size_t g = (((size_t)n * CC + i2) * HH + ys) * WW + xsrc;
        hT[p * LDH + k] = ok ? hbuf[g] : (_Float16)0.f;
        xT[p * LDX + k] = ok ? x[g] : 0.f;
    }
    __syncthreads();

    const int wave = tid >> 5;
    const int lane = tid & 31;
    const int pcol = lane & 15;     // pixel column (WMMA N)
    const int sel  = lane >> 4;     // lane-half select

    // Preload all 9 B fragments (depend only on kk) into registers.
    // B layout (32x16 f16): lane<16 holds col N=lane, K=kk*32+0..15;
    //                       lane>=16 holds col N=lane-16, K=kk*32+16..31.
    v16h Bf[KKS];
#pragma unroll
    for (int kk = 0; kk < KKS; ++kk) {
        const _Float16* bp = &hT[pcol * LDH + kk * 32 + sel * 16];
        V16 u;
        u.h[0] = *(const v8h*)(bp);
        u.h[1] = *(const v8h*)(bp + 8);
        Bf[kk] = u.v;
    }

    // A layout (16x32 f16): lane<16 holds row M=lane, K{0..15};
    //                       lane>=16 holds row M=lane-16, K{16..31}.
    const int arow  = lane & 15;
    const int acol0 = sel * 16;

    for (int oo = 0; oo < 4; ++oo) {
        const int o = wave * 4 + oo;
        const _Float16* Abase = w2h + (size_t)o * KD * KD;
        float s = 0.f;

        for (int t = 0; t < MT; ++t) {
            v8f acc = {};
            const _Float16* Ar = Abase + (size_t)(t * 16 + arow) * KD + acol0;
#pragma unroll
            for (int kk = 0; kk < KKS; ++kk) {
                V16 ua;
                ua.h[0] = *(const v8h*)(Ar + kk * 32);
                ua.h[1] = *(const v8h*)(Ar + kk * 32 + 8);
                acc = __builtin_amdgcn_wmma_f32_16x16x32_f16(
                    false, ua.v, false, Bf[kk], (short)0, acc, false, false);
            }
            // D layout: VGPR j, this lane-half -> M = sel*8 + j, N = pcol.
            const float* xp = &xT[pcol * LDX + t * 16 + sel * 8];
            float4 xa = *(const float4*)(xp);
            float4 xb = *(const float4*)(xp + 4);
            s += fast_tanh(acc[0]) * xa.x + fast_tanh(acc[1]) * xa.y +
                 fast_tanh(acc[2]) * xa.z + fast_tanh(acc[3]) * xa.w +
                 fast_tanh(acc[4]) * xb.x + fast_tanh(acc[5]) * xb.y +
                 fast_tanh(acc[6]) * xb.z + fast_tanh(acc[7]) * xb.w;
        }
        // combine the two lane halves (M 0..7 vs 8..15), then store 16 pixels
        s += __shfl_xor(s, 16, 32);
        if (lane < 16)
            out[(((size_t)n * CC + o) * HH + y) * WW + x0 + pcol] = s;
    }
}

// ---------------------------------------------------------------------------
extern "C" void kernel_launch(void* const* d_in, const int* in_sizes, int n_in,
                              void* d_out, int out_size, void* d_ws, size_t ws_size,
                              hipStream_t stream) {
    const float* x   = (const float*)d_in[0];     // [N,32,64,64]
    const float* w1  = (const float*)d_in[1];     // [32,32,3,3]
    const float* w2  = (const float*)d_in[2];     // [9216,32,3,3]
    float* out = (float*)d_out;

    const int N = in_sizes[0] / (CC * HH * WW);
    const int w2elems = in_sizes[2];              // 2,654,208

    // workspace layout
    _Float16* w2h  = (_Float16*)d_ws;                                   // 5,308,416 B
    _Float16* hbuf = (_Float16*)((char*)d_ws + (size_t)w2elems * 2);    // N*32*64*64 f16

    k_cvt_w2<<<2048, 256, 0, stream>>>(w2, w2h, w2elems);
    k_conv1<<<N * HH, 256, 0, stream>>>(x, w1, hbuf);
    k_main<<<N * HH * (WW / TP), 256, 0, stream>>>(x, w2h, hbuf, out);
}
